// SimpleNN_54228257079681
// MI455X (gfx1250) — 8-bit matrix operand order repaired and checked by reading, compiled, not run on hardware
//
#include <hip/hip_runtime.h>
#include <stdint.h>

typedef __attribute__((ext_vector_type(8))) int v8i;

#define QMAX 127.0f

__device__ __forceinline__ int q1(float v, float inv) {
    float q = rintf(v * inv);
    q = fminf(fmaxf(q, -QMAX), QMAX);
    return (int)q;
}
__device__ __forceinline__ uint32_t pack4(float a, float b, float c, float d, float inv) {
    return (uint32_t)(q1(a, inv) & 0xff)
         | ((uint32_t)(q1(b, inv) & 0xff) << 8)
         | ((uint32_t)(q1(c, inv) & 0xff) << 16)
         | ((uint32_t)(q1(d, inv) & 0xff) << 24);
}

__device__ __forceinline__ v8i loadA8(const int8_t* ap) {
    uint64_t d0 = *(const uint64_t*)(ap);
    uint64_t d1 = *(const uint64_t*)(ap + 8);
    uint64_t d2 = *(const uint64_t*)(ap + 32);
    uint64_t d3 = *(const uint64_t*)(ap + 40);
    v8i a;
    a[0] = (int)(uint32_t)d0; a[1] = (int)(d0 >> 32);
    a[2] = (int)(uint32_t)d1; a[3] = (int)(d1 >> 32);
    a[4] = (int)(uint32_t)d2; a[5] = (int)(d2 >> 32);
    a[6] = (int)(uint32_t)d3; a[7] = (int)(d3 >> 32);
    return a;
}
__device__ __forceinline__ v8i loadB8(const int8_t* wp) {
    int4 lo = *(const int4*)(wp);
    int4 hi = *(const int4*)(wp + 32);
    v8i b;
    b[0] = lo.x; b[1] = lo.y; b[2] = lo.z; b[3] = lo.w;
    b[4] = hi.x; b[5] = hi.y; b[6] = hi.z; b[7] = hi.w;
    return b;
}

#define WS_QW1 0
#define WS_SW1 212992
#define WS_QW2 214016
#define WS_SW2 218112

__global__ __launch_bounds__(256) void prep_kernel(
    const float* __restrict__ W1, const float* __restrict__ W2,
    int8_t* __restrict__ qW1, float* __restrict__ sW1,
    int8_t* __restrict__ qW2, float* __restrict__ sW2)
{
    const int t = threadIdx.x;
    {
        const float* row = W1 + (size_t)t * 784;
        float amax = 0.f;
        for (int j = 0; j < 784; j += 4) {
            float4 v = *(const float4*)(row + j);
            amax = fmaxf(amax, fmaxf(fmaxf(fabsf(v.x), fabsf(v.y)),
                                     fmaxf(fabsf(v.z), fabsf(v.w))));
        }
        float inv = amax > 0.f ? QMAX / amax : 0.f;
        sW1[t] = amax > 0.f ? amax / QMAX : 1.0f;
        int8_t* q = qW1 + (size_t)t * 832;
        for (int j = 0; j < 784; j += 4) {
            float4 v = *(const float4*)(row + j);
            *(uint32_t*)(q + j) = pack4(v.x, v.y, v.z, v.w, inv);
        }
        for (int j = 784; j < 832; j += 4) *(uint32_t*)(q + j) = 0u;
    }
    if (t < 16) {
        float amax = 0.f;
        const float* row = W2 + (size_t)t * 256;
        if (t < 10)
            for (int j = 0; j < 256; ++j) amax = fmaxf(amax, fabsf(row[j]));
        float inv = amax > 0.f ? QMAX / amax : 0.f;
        sW2[t] = (t < 10) ? (amax > 0.f ? amax / QMAX : 1.0f) : 0.f;
        int8_t* q = qW2 + t * 256;
        for (int j = 0; j < 256; j += 4) {
            if (t < 10) {
                float4 v = *(const float4*)(row + j);
                *(uint32_t*)(q + j) = pack4(v.x, v.y, v.z, v.w, inv);
            } else {
                *(uint32_t*)(q + j) = 0u;
            }
        }
    }
}

#define SMEM_BYTES 138752

__global__ __launch_bounds__(256) void mlp_kernel(
    const float* __restrict__ x,
    const float* __restrict__ b1,
    const float* __restrict__ b2,
    const int8_t* __restrict__ qW1, const float* __restrict__ sW1,
    const int8_t* __restrict__ qW2, const float* __restrict__ sW2,
    float* __restrict__ out)
{
    extern __shared__ char smem[];
    int8_t* qX   = (int8_t*)(smem);
    float*  hBuf = (float*)(smem + 53760);
    int8_t* qH   = (int8_t*)(smem + 120320);
    float*  sX   = (float*)(smem + 137216);
    float*  sH   = (float*)(smem + 137472);
    float*  pmax = (float*)(smem + 137728);

    const int t   = threadIdx.x;
    const int row = t >> 2;
    const int qq  = t & 3;
    const size_t gRow = (size_t)blockIdx.x * 64 + row;

    {
        const float* xr = x + gRow * 784 + qq * 196;
        float amax = 0.f;
        for (int j = 0; j < 196; j += 4) {
            float4 v = *(const float4*)(xr + j);
            amax = fmaxf(amax, fmaxf(fmaxf(fabsf(v.x), fabsf(v.y)),
                                     fmaxf(fabsf(v.z), fabsf(v.w))));
        }
        pmax[row * 4 + qq] = amax;
    }
    __syncthreads();

    {
        float amax = fmaxf(fmaxf(pmax[row * 4 + 0], pmax[row * 4 + 1]),
                           fmaxf(pmax[row * 4 + 2], pmax[row * 4 + 3]));
        float inv = amax > 0.f ? QMAX / amax : 0.f;
        if (qq == 0) sX[row] = amax > 0.f ? amax / QMAX : 1.0f;
        const float* xr = x + gRow * 784 + qq * 196;
        int8_t* qr = qX + row * 840 + qq * 196;
        for (int j = 0; j < 196; j += 4) {
            float4 v = *(const float4*)(xr + j);
            *(uint32_t*)(qr + j) = pack4(v.x, v.y, v.z, v.w, inv);
        }
        if (qq == 3) {
            uint32_t* pad = (uint32_t*)(qX + row * 840 + 784);
            #pragma unroll
            for (int j = 0; j < 14; ++j) pad[j] = 0u;
        }
    }
    __syncthreads();

    const int wave = t >> 5;
    const int lane = t & 31;
    const int half = lane >> 4;
    const int l16  = lane & 15;
    const int mp   = wave >> 2;
    const int nq   = wave & 3;

    v8i acc0[4], acc1[4];
    #pragma unroll
    for (int i = 0; i < 4; ++i) {
        acc0[i] = (v8i){0, 0, 0, 0, 0, 0, 0, 0};
        acc1[i] = (v8i){0, 0, 0, 0, 0, 0, 0, 0};
    }

    const int8_t* aRow0 = qX + ((2 * mp) * 16 + l16) * 840 + half * 16;
    const int8_t* aRow1 = aRow0 + 16 * 840;
    const int8_t* wBase = qW1 + (size_t)(nq * 64 + l16) * 832 + half * 16;

    for (int k = 0; k < 832; k += 64) {
        v8i a0 = loadA8(aRow0 + k);
        v8i a1 = loadA8(aRow1 + k);
        v8i b[4];
        #pragma unroll
        for (int nt = 0; nt < 4; ++nt)
            b[nt] = loadB8(wBase + (size_t)nt * 16 * 832 + k);
        #pragma unroll
        for (int nt = 0; nt < 4; ++nt) {
            acc0[nt] = __builtin_amdgcn_wmma_i32_16x16x64_iu8(
                true, a0, true, b[nt], acc0[nt], false, false);
            acc1[nt] = __builtin_amdgcn_wmma_i32_16x16x64_iu8(
                true, a1, true, b[nt], acc1[nt], false, false);
        }
    }

    {
        float sxv0[8], sxv1[8];
        #pragma unroll
        for (int v = 0; v < 8; ++v) {
            sxv0[v] = sX[(2 * mp) * 16 + v + 8 * half];
            sxv1[v] = sX[(2 * mp + 1) * 16 + v + 8 * half];
        }
        #pragma unroll
        for (int nt = 0; nt < 4; ++nt) {
            const int ncol = nq * 64 + nt * 16 + l16;
            const float sw = sW1[ncol];
            const float bb = b1[ncol];
            #pragma unroll
            for (int v = 0; v < 8; ++v) {
                const int m0 = (2 * mp) * 16 + v + 8 * half;
                const int m1 = m0 + 16;
                float v0 = (float)acc0[nt][v] * (sxv0[v] * sw) + bb;
                float v1 = (float)acc1[nt][v] * (sxv1[v] * sw) + bb;
                hBuf[m0 * 260 + ncol] = fmaxf(v0, 0.f);
                hBuf[m1 * 260 + ncol] = fmaxf(v1, 0.f);
            }
        }
    }
    __syncthreads();

    {
        const float* hr = hBuf + row * 260 + qq * 64;
        float amax = 0.f;
        for (int j = 0; j < 64; j += 4) {
            float4 v = *(const float4*)(hr + j);
            amax = fmaxf(amax, fmaxf(fmaxf(fabsf(v.x), fabsf(v.y)),
                                     fmaxf(fabsf(v.z), fabsf(v.w))));
        }
        pmax[row * 4 + qq] = amax;
    }
    __syncthreads();
    {
        float amax = fmaxf(fmaxf(pmax[row * 4 + 0], pmax[row * 4 + 1]),
                           fmaxf(pmax[row * 4 + 2], pmax[row * 4 + 3]));
        float inv = amax > 0.f ? QMAX / amax : 0.f;
        if (qq == 0) sH[row] = amax > 0.f ? amax / QMAX : 1.0f;
        const float* hr = hBuf + row * 260 + qq * 64;
        int8_t* qr = qH + row * 264 + qq * 64;
        for (int j = 0; j < 64; j += 4) {
            float4 v = *(const float4*)(hr + j);
            *(uint32_t*)(qr + j) = pack4(v.x, v.y, v.z, v.w, inv);
        }
    }
    __syncthreads();

    if (wave < 4) {
        const int m2 = wave;
        v8i acc2 = (v8i){0, 0, 0, 0, 0, 0, 0, 0};
        const int8_t* aRow2 = qH + (m2 * 16 + l16) * 264 + half * 16;
        const int8_t* wp2  = qW2 + l16 * 256 + half * 16;
        #pragma unroll
        for (int k = 0; k < 256; k += 64) {
            v8i a = loadA8(aRow2 + k);
            v8i b = loadB8(wp2 + k);
            acc2 = __builtin_amdgcn_wmma_i32_16x16x64_iu8(
                true, a, true, b, acc2, false, false);
        }
        if (l16 < 10) {
            const float sw = sW2[l16];
            const float bb = b2[l16];
            #pragma unroll
            for (int v = 0; v < 8; ++v) {
                const int m = m2 * 16 + v + 8 * half;
                float val = (float)acc2[v] * (sH[m] * sw) + bb;
                out[((size_t)blockIdx.x * 64 + m) * 10 + l16] = val;
            }
        }
    }
}

extern "C" void kernel_launch(void* const* d_in, const int* in_sizes, int n_in,
                              void* d_out, int out_size, void* d_ws, size_t ws_size,
                              hipStream_t stream) {
    const float* x  = (const float*)d_in[0];
    const float* W1 = (const float*)d_in[1];
    const float* b1 = (const float*)d_in[2];
    const float* W2 = (const float*)d_in[3];
    const float* b2 = (const float*)d_in[4];
    float* out = (float*)d_out;

    char* ws = (char*)d_ws;
    int8_t* qW1 = (int8_t*)(ws + WS_QW1);
    float*  sW1 = (float*)(ws + WS_SW1);
    int8_t* qW2 = (int8_t*)(ws + WS_QW2);
    float*  sW2 = (float*)(ws + WS_SW2);

    prep_kernel<<<1, 256, 0, stream>>>(W1, W2, qW1, sW1, qW2, sW2);

    const int nrows = in_sizes[0] / 784;
    const int nblocks = nrows / 64;
    mlp_kernel<<<nblocks, 256, SMEM_BYTES, stream>>>(
        x, b1, b2, qW1, sW1, qW2, sW2, out);
}
